// GroupedQueryAttention_18571438588604
// MI455X (gfx1250) — compile-verified, matrix operand order checked by reading, not run on hardware
//
#include <hip/hip_runtime.h>

typedef __bf16 bf16_t;
typedef __attribute__((ext_vector_type(16))) __bf16 v16bf;
typedef __attribute__((ext_vector_type(8)))  float  v8f;

#define S_LEN  2048
#define E_DIM  2048
#define NQ_H   16
#define NKV_H  4
#define HD_DIM 128
#define WIN    512
#define QKV_N  3072

__device__ __forceinline__ v8f wmma_bf16(v16bf a, v16bf b, v8f c) {
  // D = A(16x32 bf16) * B(32x16 bf16) + C(16x16 f32)
  return __builtin_amdgcn_wmma_f32_16x16x32_bf16(false, a, false, b, (short)0, c, false, false);
}

// CDNA5 async copy: global -> LDS, 16B per lane, tracked by ASYNCcnt.
__device__ __forceinline__ void async_b128(unsigned lds_addr, unsigned byte_off,
                                           const void* base) {
  asm volatile("global_load_async_to_lds_b128 %0, %1, %2"
               :: "v"(lds_addr), "v"(byte_off), "s"(base) : "memory");
}
template <int N>
__device__ __forceinline__ void wait_async() {
  asm volatile("s_wait_asynccnt %0" :: "i"(N) : "memory");
}
__device__ __forceinline__ unsigned lds_off(const void* p) {
  // generic pointer to LDS: low 32 bits are the LDS byte address
  return (unsigned)(uintptr_t)p;
}

// ---------------------------------------------------------------------------
// Packing / conversion kernels
// ---------------------------------------------------------------------------
__global__ void f32_to_bf16_kernel(const float* __restrict__ src,
                                   bf16_t* __restrict__ dst, long n) {
  long i = (long)blockIdx.x * blockDim.x + threadIdx.x;
  if (i < n) dst[i] = (bf16_t)src[i];
}

__global__ void pack_bqkv_kernel(const float* __restrict__ bq,
                                 const float* __restrict__ bk,
                                 const float* __restrict__ bv,
                                 float* __restrict__ out) {
  int c = blockIdx.x * blockDim.x + threadIdx.x;
  if (c >= QKV_N) return;
  out[c] = (c < 2048) ? bq[c] : (c < 2560) ? bk[c - 2048] : bv[c - 2560];
}

// Tiled transpose + bf16 pack: out[n][k] = src[k][n], src = (Wq | Wk | Wv)
// column-concat if Wk != nullptr, else a single matrix Wq of width N.
__global__ __launch_bounds__(256)
void transpose_pack_kernel(const float* __restrict__ Wq,
                           const float* __restrict__ Wk,
                           const float* __restrict__ Wv,
                           bf16_t* __restrict__ out, int K, int N) {
  __shared__ float tile[64][65];
  const int tx = blockIdx.x * 64;          // n range
  const int ty = blockIdx.y * 64;          // k range
  const int tid = threadIdx.x;
  {
    const int c = tid & 63, r0 = tid >> 6;
#pragma unroll
    for (int i = 0; i < 16; ++i) {
      const int r = r0 + 4 * i;
      const int kk = ty + r, nn = tx + c;
      float v;
      if (Wk == nullptr)      v = Wq[(size_t)kk * N + nn];
      else if (nn < 2048)     v = Wq[(size_t)kk * 2048 + nn];
      else if (nn < 2560)     v = Wk[(size_t)kk * 512 + (nn - 2048)];
      else                    v = Wv[(size_t)kk * 512 + (nn - 2560)];
      tile[r][c] = v;
    }
  }
  __syncthreads();
  {
    const int k2 = tid & 63, n0 = tid >> 6;
#pragma unroll
    for (int i = 0; i < 16; ++i) {
      const int n2 = n0 + 4 * i;
      out[(size_t)(tx + n2) * K + ty + k2] = (bf16_t)tile[k2][n2];
    }
  }
}

// ---------------------------------------------------------------------------
// Tiled bf16 GEMM: C(MxN) = A(MxK,row) * Bt(NxK,row = B^T) + bias, via WMMA.
// BM=BN=128, BK=32; 8 waves, each computes 64x32 (4x2 wmma tiles).
// Both tiles staged with async global->LDS b128 copies, double buffered.
// ---------------------------------------------------------------------------
template <bool OUT_BF16>
__global__ __launch_bounds__(256)
void gemm_bf16_kernel(const bf16_t* __restrict__ A, const bf16_t* __restrict__ Bt,
                      const float* __restrict__ bias, void* __restrict__ C,
                      int M, int N, int K) {
  __shared__ alignas(32) bf16_t As[2][128][48];      // [m][k], padded
  __shared__ alignas(32) bf16_t Bs[2][128][48];      // [n][k], padded

  const int tid  = threadIdx.x;
  const int wave = tid >> 5;
  const int lane = tid & 31;
  const int hl   = lane >> 4;
  const int lx   = lane & 15;
  const int bm   = blockIdx.y * 128;
  const int bn   = blockIdx.x * 128;
  const int wm   = (wave >> 2) * 64;   // 0 / 64
  const int wn   = (wave & 3) * 32;    // 0..96

  v8f acc[4][2] = {};

  auto issue_tile = [&](int buf, int k0) {
    const int r = tid >> 2;             // 0..63
    const int seg = (tid & 3) * 8;      // 8 bf16 = 16B chunk
#pragma unroll
    for (int i = 0; i < 2; ++i) {
      const int row = r + 64 * i;
      async_b128(lds_off(&As[buf][row][seg]),
                 (unsigned)(((size_t)(bm + row) * K + k0 + seg) * sizeof(bf16_t)), A);
      async_b128(lds_off(&Bs[buf][row][seg]),
                 (unsigned)(((size_t)(bn + row) * K + k0 + seg) * sizeof(bf16_t)), Bt);
    }
  };

  issue_tile(0, 0);
  wait_async<0>();
  __syncthreads();

  const int kTiles = K / 32;
  for (int kt = 0; kt < kTiles; ++kt) {
    const int buf = kt & 1;
    if (kt + 1 < kTiles) issue_tile(buf ^ 1, (kt + 1) * 32);

    v16bf a[4], bb[2];
#pragma unroll
    for (int mi = 0; mi < 4; ++mi)
      a[mi] = *(const v16bf*)&As[buf][wm + mi * 16 + lx][hl * 16];
#pragma unroll
    for (int ni = 0; ni < 2; ++ni)
      bb[ni] = *(const v16bf*)&Bs[buf][wn + ni * 16 + lx][hl * 16];
#pragma unroll
    for (int mi = 0; mi < 4; ++mi)
#pragma unroll
      for (int ni = 0; ni < 2; ++ni)
        acc[mi][ni] = wmma_bf16(a[mi], bb[ni], acc[mi][ni]);

    wait_async<0>();
    __syncthreads();
  }

  // Epilogue: C row = vgpr_r + 8*(lane/16), col = lane%16 per 16x16 tile
#pragma unroll
  for (int ni = 0; ni < 2; ++ni) {
    const int n = bn + wn + ni * 16 + lx;
    const float bv = bias ? bias[n] : 0.f;
#pragma unroll
    for (int mi = 0; mi < 4; ++mi) {
#pragma unroll
      for (int r = 0; r < 8; ++r) {
        const int m = bm + wm + mi * 16 + hl * 8 + r;
        const float val = acc[mi][ni][r] + bv;
        if (OUT_BF16) ((bf16_t*)C)[(size_t)m * N + n] = (bf16_t)val;
        else          ((float*)C)[(size_t)m * N + n]  = val;
      }
    }
  }
}

// ---------------------------------------------------------------------------
// RoPE + scatter.  qkv: [B*S][3072] bf16 ->
//   q [B][NQ][S][HD],  k [B][NKV][S][HD],  v TRANSPOSED: [B][NKV][HD][S]
// ---------------------------------------------------------------------------
__global__ void rope_scatter_kernel(const bf16_t* __restrict__ qkv,
                                    bf16_t* __restrict__ q,
                                    bf16_t* __restrict__ k,
                                    bf16_t* __restrict__ v) {
  long idx = (long)blockIdx.x * blockDim.x + threadIdx.x;
  if (idx >= (long)2 * S_LEN * 24 * 64) return;
  int pair = (int)(idx & 63);
  long t = idx >> 6;
  int hh = (int)(t % 24); t /= 24;
  int s  = (int)(t % S_LEN);
  int b  = (int)(t / S_LEN);

  const bf16_t* src = qkv + (size_t)(b * S_LEN + s) * QKV_N;
  float inv_freq = __powf(10000.0f, -(2.0f * pair) / 128.0f);
  float sn, cs;
  __sincosf((float)s * inv_freq, &sn, &cs);

  if (hh < NQ_H) {
    float xe = (float)src[hh * HD_DIM + 2 * pair];
    float xo = (float)src[hh * HD_DIM + 2 * pair + 1];
    size_t d = (((size_t)b * NQ_H + hh) * S_LEN + s) * HD_DIM + 2 * pair;
    q[d]     = (bf16_t)(xe * cs - xo * sn);
    q[d + 1] = (bf16_t)(xe * sn + xo * cs);
  } else if (hh < NQ_H + NKV_H) {
    int h = hh - NQ_H;
    float xe = (float)src[2048 + h * HD_DIM + 2 * pair];
    float xo = (float)src[2048 + h * HD_DIM + 2 * pair + 1];
    size_t d = (((size_t)b * NKV_H + h) * S_LEN + s) * HD_DIM + 2 * pair;
    k[d]     = (bf16_t)(xe * cs - xo * sn);
    k[d + 1] = (bf16_t)(xe * sn + xo * cs);
  } else {
    int h = hh - NQ_H - NKV_H;
    size_t sidx = 2560 + h * HD_DIM + 2 * pair;
    // transposed: v[b][h][hd][s]
    size_t d = (((size_t)b * NKV_H + h) * HD_DIM + 2 * pair) * S_LEN + s;
    v[d]         = src[sidx];
    v[d + S_LEN] = src[sidx + 1];
  }
}

// ---------------------------------------------------------------------------
// Sliding-window flash attention. Block = (128 queries, 1 q-head, 1 batch),
// 8 waves x 16 query rows, key blocks of 64. K and V(pre-transposed) blocks
// async-staged into LDS with split waits so the V transfer overlaps softmax.
// Output: attn [B*S][NQ*HD] bf16.
// ---------------------------------------------------------------------------
__global__ __launch_bounds__(256)
void attn_kernel(const bf16_t* __restrict__ q, const bf16_t* __restrict__ k,
                 const bf16_t* __restrict__ v, bf16_t* __restrict__ o) {
  constexpr int KB = 64;
  __shared__ alignas(32) bf16_t Ks[KB][HD_DIM + 16];   // [key][hd]
  __shared__ alignas(32) bf16_t Vt[HD_DIM][KB + 16];   // [hd][key]
  __shared__ alignas(32) bf16_t Ps[8][16][KB + 16];    // per-wave P scratch

  const int tid  = threadIdx.x;
  const int wave = tid >> 5;
  const int lane = tid & 31;
  const int hl   = lane >> 4;
  const int lx   = lane & 15;

  const int qt = blockIdx.x, h = blockIdx.y, b = blockIdx.z;
  const int kvh   = h >> 2;             // NQ/NKV = 4
  const int qbase = qt * 128;

  const bf16_t* qp  = q + ((size_t)b * NQ_H  + h)   * S_LEN * HD_DIM;
  const bf16_t* kp  = k + ((size_t)b * NKV_H + kvh) * S_LEN * HD_DIM;
  const bf16_t* vpT = v + ((size_t)b * NKV_H + kvh) * HD_DIM * S_LEN;

  // Q fragments for this wave's 16 rows (A operand: contiguous 16 bf16/lane)
  v16bf qf[4];
#pragma unroll
  for (int kc = 0; kc < 4; ++kc)
    qf[kc] = *(const v16bf*)(qp + (size_t)(qbase + wave * 16 + lx) * HD_DIM +
                             kc * 32 + hl * 16);

  v8f O[8] = {};
  float mrow[8], lrow[8];
#pragma unroll
  for (int r = 0; r < 8; ++r) { mrow[r] = -__builtin_inff(); lrow[r] = 0.f; }

  int kb0 = (qbase - WIN) / KB; if (kb0 < 0) kb0 = 0;
  const int kb1 = (qbase + 127) / KB;
  const float scale = 0.08838834764831845f;  // 1/sqrt(128)

  for (int kb = kb0; kb <= kb1; ++kb) {
    const int kbase = kb * KB;

    // async stage: all K chunks first, then all V chunks (in-order completion
    // lets "asynccnt <= 4" mean: K block resident, V still in flight).
#pragma unroll
    for (int i = 0; i < 4; ++i) {
      const int idx = i * 256 + tid;
      const int kr = idx >> 4, ks = (idx & 15) * 8;   // K: 16 x 16B chunks/row
      async_b128(lds_off(&Ks[kr][ks]),
                 (unsigned)(((size_t)(kbase + kr) * HD_DIM + ks) * sizeof(bf16_t)), kp);
    }
#pragma unroll
    for (int i = 0; i < 4; ++i) {
      const int idx = i * 256 + tid;
      const int hr = idx >> 3, vs = (idx & 7) * 8;    // V: 8 x 16B chunks/row
      async_b128(lds_off(&Vt[hr][vs]),
                 (unsigned)(((size_t)hr * S_LEN + kbase + vs) * sizeof(bf16_t)), vpT);
    }
    wait_async<4>();          // K done; V overlaps scores+softmax below
    __syncthreads();

    // scores = q @ k^T  (4 key tiles of 16, K-dim = HD = 4 chunks of 32)
    v8f sc[4] = {};
#pragma unroll
    for (int nt = 0; nt < 4; ++nt)
#pragma unroll
      for (int kc = 0; kc < 4; ++kc) {
        v16bf kf = *(const v16bf*)&Ks[nt * 16 + lx][kc * 32 + hl * 16];
        sc[nt] = wmma_bf16(qf[kc], kf, sc[nt]);
      }

    // mask + online softmax (in place in sc; p written straight to Ps as bf16)
#pragma unroll
    for (int r = 0; r < 8; ++r) {
      const int i = qbase + wave * 16 + hl * 8 + r;
      float rm = -__builtin_inff();
#pragma unroll
      for (int nt = 0; nt < 4; ++nt) {
        const int j = kbase + nt * 16 + lx;
        float s = sc[nt][r] * scale;
        const bool ok = (j <= i) && (i - j <= WIN);
        s = ok ? s : -__builtin_inff();
        sc[nt][r] = s;
        rm = fmaxf(rm, s);
      }
#pragma unroll
      for (int d = 1; d < 16; d <<= 1) rm = fmaxf(rm, __shfl_xor(rm, d, 32));
      const float mnew = fmaxf(mrow[r], rm);
      float alpha, rsum = 0.f;
      if (mnew == -__builtin_inff()) {   // whole block masked for this row
        alpha = 1.f;
#pragma unroll
        for (int nt = 0; nt < 4; ++nt)
          Ps[wave][hl * 8 + r][nt * 16 + lx] = (bf16_t)0.f;
      } else {
        alpha = __expf(mrow[r] - mnew);
#pragma unroll
        for (int nt = 0; nt < 4; ++nt) {
          const float p = __expf(sc[nt][r] - mnew);
          rsum += p;
          Ps[wave][hl * 8 + r][nt * 16 + lx] = (bf16_t)p;
        }
      }
      mrow[r] = mnew;
#pragma unroll
      for (int d = 1; d < 16; d <<= 1) rsum += __shfl_xor(rsum, d, 32);
      lrow[r] = lrow[r] * alpha + rsum;
#pragma unroll
      for (int ht = 0; ht < 8; ++ht) O[ht][r] *= alpha;
    }

    asm volatile("s_wait_dscnt 0x0" ::: "memory");
    wait_async<0>();          // V block now resident
    __syncthreads();

    // O += P @ V  (8 hd tiles, key dim = 64 = 2 chunks of 32)
#pragma unroll
    for (int ht = 0; ht < 8; ++ht)
#pragma unroll
      for (int kc = 0; kc < 2; ++kc) {
        v16bf pf = *(const v16bf*)&Ps[wave][lx][kc * 32 + hl * 16];
        v16bf vf = *(const v16bf*)&Vt[ht * 16 + lx][kc * 32 + hl * 16];
        O[ht] = wmma_bf16(pf, vf, O[ht]);
      }
    __syncthreads();   // all waves done with Ks/Vt before next async overwrite
  }

  // normalize + store: attn[(b*S + row)][h*HD + col]
#pragma unroll
  for (int ht = 0; ht < 8; ++ht) {
#pragma unroll
    for (int r = 0; r < 8; ++r) {
      const int row = qbase + wave * 16 + hl * 8 + r;
      const float val = O[ht][r] / lrow[r];
      o[((size_t)b * S_LEN + row) * (NQ_H * HD_DIM) + h * HD_DIM + ht * 16 + lx] =
          (bf16_t)val;
    }
  }
}

// ---------------------------------------------------------------------------
// Launch
// ---------------------------------------------------------------------------
extern "C" void kernel_launch(void* const* d_in, const int* in_sizes, int n_in,
                              void* d_out, int out_size, void* d_ws, size_t ws_size,
                              hipStream_t stream) {
  const float* x  = (const float*)d_in[0];
  const float* Wq = (const float*)d_in[1];
  const float* bq = (const float*)d_in[2];
  const float* Wk = (const float*)d_in[3];
  const float* bk = (const float*)d_in[4];
  const float* Wv = (const float*)d_in[5];
  const float* bv = (const float*)d_in[6];
  const float* Wo = (const float*)d_in[7];
  const float* bo = (const float*)d_in[8];
  float* out = (float*)d_out;

  char* ws = (char*)d_ws;
  size_t off = 0;
  auto alloc = [&](size_t bytes) -> void* {
    void* p = ws + off;
    off = (off + bytes + 255) & ~(size_t)255;
    return p;
  };

  const size_t M = 2 * S_LEN;                                // 4096
  bf16_t* xb     = (bf16_t*)alloc(M * E_DIM * 2);            // x bf16 [M][K]
  bf16_t* wqkvT  = (bf16_t*)alloc((size_t)QKV_N * E_DIM * 2);// [3072][2048]
  float*  bqkv   = (float*)alloc(QKV_N * 4);
  bf16_t* woT    = (bf16_t*)alloc((size_t)E_DIM * E_DIM * 2);// [2048][2048]
  bf16_t* qkvout = (bf16_t*)alloc(M * QKV_N * 2);
  bf16_t* qh     = (bf16_t*)alloc((size_t)2 * NQ_H  * S_LEN * HD_DIM * 2);
  bf16_t* kh     = (bf16_t*)alloc((size_t)2 * NKV_H * S_LEN * HD_DIM * 2);
  bf16_t* vhT    = (bf16_t*)alloc((size_t)2 * NKV_H * HD_DIM * S_LEN * 2);
  bf16_t* attn   = (bf16_t*)alloc(M * E_DIM * 2);

  // 1) pack inputs to bf16 (weights transposed for contiguous GEMM B tiles)
  {
    long n = (long)M * E_DIM;
    f32_to_bf16_kernel<<<(n + 255) / 256, 256, 0, stream>>>(x, xb, n);
  }
  transpose_pack_kernel<<<dim3(QKV_N / 64, E_DIM / 64), 256, 0, stream>>>(
      Wq, Wk, Wv, wqkvT, E_DIM, QKV_N);
  pack_bqkv_kernel<<<(QKV_N + 255) / 256, 256, 0, stream>>>(bq, bk, bv, bqkv);
  transpose_pack_kernel<<<dim3(E_DIM / 64, E_DIM / 64), 256, 0, stream>>>(
      Wo, nullptr, nullptr, woT, E_DIM, E_DIM);

  // 2) QKV projection GEMM: [4096 x 2048] * [2048 x 3072] + bias -> bf16
  gemm_bf16_kernel<true><<<dim3(QKV_N / 128, M / 128), 256, 0, stream>>>(
      xb, wqkvT, bqkv, qkvout, (int)M, QKV_N, E_DIM);

  // 3) RoPE + scatter to head-major (V transposed)
  {
    long n = (long)2 * S_LEN * 24 * 64;
    rope_scatter_kernel<<<(n + 255) / 256, 256, 0, stream>>>(qkvout, qh, kh, vhT);
  }

  // 4) sliding-window attention
  attn_kernel<<<dim3(S_LEN / 128, NQ_H, 2), 256, 0, stream>>>(qh, kh, vhT, attn);

  // 5) output projection GEMM: [4096 x 2048] * [2048 x 2048] + bo -> f32 d_out
  gemm_bf16_kernel<false><<<dim3(E_DIM / 128, M / 128), 256, 0, stream>>>(
      attn, woT, bo, out, (int)M, E_DIM, E_DIM);
}
